// PieceWiseLinearIntLayer_19284403159597
// MI455X (gfx1250) — matrix operand order repaired and checked by reading, not compiled, not run on hardware
//
#include <hip/hip_runtime.h>
#include <hip/hip_bf16.h>
#include <math.h>

// ---------------------------------------------------------------------------
// Fused PieceWiseLinearIntLayer for MI455X (gfx1250, wave32, WMMA)
//   b_out = gelu(x) @ W^T + bias           (split-bf16 WMMA, 3-product f32-ish)
//   lin   = seg_x @ slopes0^T + seg_cnt @ icpt0^T
//   out   = prelu(layernorm(b_out + lin))
// ---------------------------------------------------------------------------

typedef float  v8f   __attribute__((ext_vector_type(8)));
typedef __bf16 v16bf __attribute__((ext_vector_type(16)));

#define N_IN   1024
#define N_OUT  1024
#define BATCH  8192
#define NSEG   9

struct Frag32B { uint4 a, b; };

static __device__ __forceinline__ v16bf load_frag16(const uint4* p) {
  Frag32B f;
  f.a = p[0];
  f.b = p[1];
  return __builtin_bit_cast(v16bf, f);
}

static __device__ __forceinline__ unsigned int pack_bf2(float x0, float x1) {
  unsigned short b0 = __builtin_bit_cast(unsigned short, (__bf16)x0);
  unsigned short b1 = __builtin_bit_cast(unsigned short, (__bf16)x1);
  return ((unsigned int)b1 << 16) | (unsigned int)b0;
}
static __device__ __forceinline__ float bf16val(float x) { return (float)(__bf16)x; }
static __device__ __forceinline__ float gelu_exact(float x) {
  return 0.5f * x * (1.0f + erff(x * 0.70710678118654752f));
}

// ---------------------------------------------------------------------------
// Kernel 1: pack W (1024x1024 f32, row-major [n][k]) into bf16 hi/lo
// B-fragments matching the 16-bit 32x16 WMMA B VGPR layout:
//   lane L<16 : N = nt*16+L,    V0..3 K={0..7},  V4..7 K={16..23}  (+kt*32)
//   lane L>=16: N = nt*16+L-16, V0..3 K={8..15}, V4..7 K={24..31}  (+kt*32)
// frag id f = kt*64 + nt ; storage: dword index f*256 + lane*8 + v
// ---------------------------------------------------------------------------
__global__ __launch_bounds__(256) void pack_w(const float* __restrict__ W,
                                              unsigned int* __restrict__ Bhi,
                                              unsigned int* __restrict__ Blo) {
  int g    = blockIdx.x * 256 + threadIdx.x;
  int frag = g >> 5, lane = g & 31;
  int kt   = frag >> 6, nt = frag & 63;
  int half = (lane >= 16);
  int N    = nt * 16 + (lane & 15);
  const float* wr = W + (size_t)N * N_IN;
  size_t base = (size_t)frag * 256 + (size_t)lane * 8;
#pragma unroll
  for (int v = 0; v < 8; ++v) {
    int k0 = kt * 32 + ((v < 4) ? 2 * v : 16 + 2 * (v - 4)) + (half ? 8 : 0);
    float w0 = wr[k0], w1 = wr[k0 + 1];
    float h0 = bf16val(w0), h1 = bf16val(w1);
    Bhi[base + v] = pack_bf2(w0, w1);
    Blo[base + v] = pack_bf2(w0 - h0, w1 - h1);
  }
}

// ---------------------------------------------------------------------------
// Kernel 2: pack gelu(x) into bf16 hi/lo A-fragments (16-bit 16x32 A layout):
//   lane L<16 : M=mt*16+L,    V0..3 K={0,1..7}, V4..7 K={16..23}
//   lane L>=16: M=mt*16+L-16, V0..3 K={8..15},  V4..7 K={24..31}   (+kt*32)
// frag id f = mt*32 + kt
// ---------------------------------------------------------------------------
__global__ __launch_bounds__(256) void pack_a(const float* __restrict__ X,
                                              unsigned int* __restrict__ Ahi,
                                              unsigned int* __restrict__ Alo) {
  int g    = blockIdx.x * 256 + threadIdx.x;
  int frag = g >> 5, lane = g & 31;
  int mt   = frag >> 5, kt = frag & 31;
  int half = (lane >= 16);
  int M    = mt * 16 + (lane & 15);
  const float* xr = X + (size_t)M * N_IN;
  size_t base = (size_t)frag * 256 + (size_t)lane * 8;
#pragma unroll
  for (int v = 0; v < 8; ++v) {
    int k0 = kt * 32 + ((v < 4) ? 2 * v : 16 + 2 * (v - 4)) + (half ? 8 : 0);
    float g0 = gelu_exact(xr[k0]);
    float g1 = gelu_exact(xr[k0 + 1]);
    float h0 = bf16val(g0), h1 = bf16val(g1);
    Ahi[base + v] = pack_bf2(g0, g1);
    Alo[base + v] = pack_bf2(g0 - h0, g1 - h1);
  }
}

// ---------------------------------------------------------------------------
// Kernel 3: per-row 9-bin segment stats. One wave32 per batch row.
// SEG[row*18 + s]     = sum of x falling in segment s
// SEG[row*18 + 9 + s] = count of x falling in segment s
// ---------------------------------------------------------------------------
__global__ __launch_bounds__(256) void seg_stats(const float* __restrict__ X,
                                                 float* __restrict__ SEG) {
  int row  = blockIdx.x * 8 + (threadIdx.x >> 5);
  int lane = threadIdx.x & 31;
  const float* xr = X + (size_t)row * N_IN;
  float sx[9], sc[9];
#pragma unroll
  for (int s = 0; s < 9; ++s) { sx[s] = 0.f; sc[s] = 0.f; }
  for (int j = 0; j < 32; ++j) {
    float xv = xr[j * 32 + lane];
    int cnt = 0;
#pragma unroll
    for (int gg = 0; gg < 10; ++gg)
      cnt += (xv >= (-1.0f + (2.0f / 9.0f) * (float)gg)) ? 1 : 0;
    int sb = min(max(cnt - 1, 0), 8);
#pragma unroll
    for (int s = 0; s < 9; ++s) {           // branchless, register-resident bins
      float m = (sb == s) ? 1.f : 0.f;
      sx[s] += m * xv;
      sc[s] += m;
    }
  }
#pragma unroll
  for (int s = 0; s < 9; ++s) {
#pragma unroll
    for (int m = 16; m >= 1; m >>= 1) {      // wave32 butterfly
      sx[s] += __shfl_xor(sx[s], m, 32);
      sc[s] += __shfl_xor(sc[s], m, 32);
    }
  }
  if (lane == 0) {
#pragma unroll
    for (int s = 0; s < 9; ++s) {
      SEG[(size_t)row * 18 + s]     = sx[s];
      SEG[(size_t)row * 18 + 9 + s] = sc[s];
    }
  }
}

// ---------------------------------------------------------------------------
// Kernel 4: fused GEMM (split-bf16 WMMA) + bias + piecewise-linear correction
//           + LayerNorm + PReLU.
// One workgroup (8 waves) per 16 batch rows; wave w owns columns [w*128,w*128+128).
// ---------------------------------------------------------------------------
__global__ __launch_bounds__(256) void fused_main(
    const uint4* __restrict__ Ahi4, const uint4* __restrict__ Alo4,
    const uint4* __restrict__ Bhi4, const uint4* __restrict__ Blo4,
    const float* __restrict__ SEG,
    const float* __restrict__ biases,
    const float* __restrict__ slopes,     // (1024,1024,9): slopes0[o][s] = slopes[o*9216+s]
    const float* __restrict__ icpts,
    const float* __restrict__ prelu,
    float* __restrict__ out) {
  __shared__ float s_seg[16][18];
  __shared__ float s_red[16][2];

  const int blk  = blockIdx.x;           // 0..511 : rows blk*16 .. blk*16+15
  const int wave = threadIdx.x >> 5;     // 0..7
  const int lane = threadIdx.x & 31;

  for (int i = threadIdx.x; i < 16 * 18; i += 256)
    ((float*)s_seg)[i] = SEG[(size_t)blk * (16 * 18) + i];
  if (threadIdx.x < 32) ((float*)s_red)[threadIdx.x] = 0.f;
  __syncthreads();

  v8f acc[8] = {};                       // 8 tiles of 16x16 f32 (C/D layout)
  const int lq = lane * 2;               // lane offset in uint4 units (8 dwords)

  for (int kt = 0; kt < 32; ++kt) {
    size_t af = ((size_t)(blk * 32 + kt)) * 64 + lq;
    v16bf ahi = load_frag16(Ahi4 + af);
    v16bf alo = load_frag16(Alo4 + af);
    if (kt + 1 < 32)                     // hint next A fragment -> global_prefetch_b8
      __builtin_prefetch(Ahi4 + ((size_t)(blk * 32 + kt + 1)) * 64 + lq, 0, 1);
#pragma unroll
    for (int t = 0; t < 8; ++t) {
      int nt = wave * 8 + t;
      size_t fb = ((size_t)(kt * 64 + nt)) * 64 + lq;
      v16bf bhi = load_frag16(Bhi4 + fb);
      v16bf blo = load_frag16(Blo4 + fb);
      // A*B ~= Ahi*Bhi + Ahi*Blo + Alo*Bhi  (near-f32 accuracy in bf16 WMMA)
      acc[t] = __builtin_amdgcn_wmma_f32_16x16x32_bf16(false, ahi, false, bhi,
                                                       (short)0, acc[t], false, false);
      acc[t] = __builtin_amdgcn_wmma_f32_16x16x32_bf16(false, ahi, false, blo,
                                                       (short)0, acc[t], false, false);
      acc[t] = __builtin_amdgcn_wmma_f32_16x16x32_bf16(false, alo, false, bhi,
                                                       (short)0, acc[t], false, false);
    }
  }

  // C/D layout: VGPR r, lanes 0-15 -> (M=r,   N=lane),
  //                     lanes 16-31 -> (M=r+8, N=lane-16)
  const int mbase = (lane < 16) ? 0 : 8;
  int col[8];
#pragma unroll
  for (int t = 0; t < 8; ++t) col[t] = wave * 128 + t * 16 + (lane & 15);

  // bias + rank-18 piecewise-linear correction
#pragma unroll
  for (int t = 0; t < 8; ++t) {
    float bv = biases[col[t]];
    const float* sl = slopes + (size_t)col[t] * (N_IN * NSEG);
    const float* ic = icpts  + (size_t)col[t] * (N_IN * NSEG);
    float slv[9], icv[9];
#pragma unroll
    for (int s = 0; s < 9; ++s) { slv[s] = sl[s]; icv[s] = ic[s]; }
#pragma unroll
    for (int r = 0; r < 8; ++r) {
      int m = mbase + r;
      float lin = bv;
#pragma unroll
      for (int s = 0; s < 9; ++s)
        lin += s_seg[m][s] * slv[s] + s_seg[m][9 + s] * icv[s];
      acc[t][r] += lin;
    }
  }

  // LayerNorm statistics: reduce across tiles, then across the 16-lane half
  // (all lanes of a half share the same row set), then across waves via LDS.
#pragma unroll
  for (int r = 0; r < 8; ++r) {
    float v = 0.f, v2 = 0.f;
#pragma unroll
    for (int t = 0; t < 8; ++t) { float a = acc[t][r]; v += a; v2 += a * a; }
#pragma unroll
    for (int m = 8; m >= 1; m >>= 1) {
      v  += __shfl_xor(v,  m, 16);
      v2 += __shfl_xor(v2, m, 16);
    }
    if ((lane & 15) == r) {              // one lane per half commits this row
      atomicAdd(&s_red[mbase + r][0], v);   // ds_add_f32
      atomicAdd(&s_red[mbase + r][1], v2);
    }
  }
  __syncthreads();

  const float p = prelu[0];
#pragma unroll
  for (int r = 0; r < 8; ++r) {
    int m = mbase + r;
    float mu  = s_red[m][0] * (1.0f / (float)N_OUT);
    float var = s_red[m][1] * (1.0f / (float)N_OUT) - mu * mu;
    float rs  = rsqrtf(var + 1e-5f);
    size_t rowoff = ((size_t)(blk * 16 + m)) * N_OUT;
#pragma unroll
    for (int t = 0; t < 8; ++t) {
      float yn = (acc[t][r] - mu) * rs;
      out[rowoff + col[t]] = (yn >= 0.f) ? yn : p * yn;
    }
  }
}

// ---------------------------------------------------------------------------
// Host launcher. Workspace layout (dwords):
//   Bhi: 2048*256   Blo: 2048*256   Ahi: 16384*256   Alo: 16384*256
//   SEG: 8192*18 floats                     (total ~36.6 MB)
// ---------------------------------------------------------------------------
extern "C" void kernel_launch(void* const* d_in, const int* in_sizes, int n_in,
                              void* d_out, int out_size, void* d_ws, size_t ws_size,
                              hipStream_t stream) {
  (void)in_sizes; (void)n_in; (void)out_size; (void)ws_size;
  const float* x  = (const float*)d_in[0];
  const float* w  = (const float*)d_in[1];
  const float* b  = (const float*)d_in[2];
  const float* sl = (const float*)d_in[3];
  const float* ic = (const float*)d_in[4];
  const float* pw = (const float*)d_in[5];

  unsigned int* Bhi = (unsigned int*)d_ws;
  unsigned int* Blo = Bhi + (size_t)2048 * 256;
  unsigned int* Ahi = Blo + (size_t)2048 * 256;
  unsigned int* Alo = Ahi + (size_t)16384 * 256;
  float*        SEG = (float*)(Alo + (size_t)16384 * 256);
  float*        out = (float*)d_out;

  pack_w   <<<256,  256, 0, stream>>>(w, Bhi, Blo);
  pack_a   <<<2048, 256, 0, stream>>>(x, Ahi, Alo);
  seg_stats<<<1024, 256, 0, stream>>>(x, SEG);
  fused_main<<<512, 256, 0, stream>>>((const uint4*)Ahi, (const uint4*)Alo,
                                      (const uint4*)Bhi, (const uint4*)Blo,
                                      SEG, b, sl, ic, pw, out);
}
